// IKeypoint_1460288880851
// MI455X (gfx1250) — matrix operand order repaired and checked by reading, not compiled, not run on hardware
//
#include <hip/hip_runtime.h>

typedef __attribute__((ext_vector_type(16))) __bf16 v16bf;
typedef __attribute__((ext_vector_type(8)))  float  v8f;
typedef __attribute__((ext_vector_type(4)))  int    v4i_;

#define NT      11      // N tiles of 16 -> 176 >= 171 real outputs
#define OPAD    176
#define NOUT    171     // 18 det + 153 kpt channels
#define NO      57      // outputs per anchor row
#define TOTROWS 25200   // 3*(6400+1600+400)

// ---- CDNA5 async global->LDS path (guarded; fallback = staged ds path) ----
#if defined(__AMDGCN__)
#define HAVE_ASYNC_LDS __has_builtin(__builtin_amdgcn_global_load_async_to_lds_b128)
#else
#define HAVE_ASYNC_LDS 0
#endif

#if HAVE_ASYNC_LDS
// param0: int4 addrspace(1)* (printed as "int4 __device__*"), param1: LDS side
#define ASYNC_CP16(g, l)                                            \
  __builtin_amdgcn_global_load_async_to_lds_b128(                   \
      (__attribute__((address_space(1))) v4i_*)(g),                 \
      (__attribute__((address_space(3))) v4i_*)(l), 0, 0)
#endif

__device__ __forceinline__ void wait_dscnt0() {
  asm volatile("s_wait_dscnt 0x0" ::: "memory");
}
__device__ __forceinline__ void wait_async_le4() {
#if HAVE_ASYNC_LDS
  asm volatile("s_wait_asynccnt 0x4" ::: "memory");
#endif
}
__device__ __forceinline__ void wait_async0() {
#if HAVE_ASYNC_LDS
  asm volatile("s_wait_asynccnt 0x0" ::: "memory");
#endif
}

__device__ __forceinline__ float fsigmoid(float x) {
  float e = __builtin_amdgcn_exp2f(x * -1.44269504088896340736f);
  return __builtin_amdgcn_rcpf(1.0f + e);
}

// ---------------------------------------------------------------------------
// Prep: fold im into det weights, split f32 -> bf16 hi/lo, and lay the result
// out in this kernel's per-lane B-fragment k order (32x16 K x N, contiguous k):
//   lane = n + (k>=16 ? 16 : 0), element e = k & 15, value = W[o=16t+n][c]
// Buffer layout (bf16): [ks][t][{hi,lo}][lane(32)][e(16)]
// ---------------------------------------------------------------------------
__global__ void pack_weights(const float* __restrict__ w,
                             const float* __restrict__ wk,
                             const float* __restrict__ im,
                             __bf16* __restrict__ dst, int C) {
  int idx = blockIdx.x * 256 + threadIdx.x;
  if (idx >= OPAD * C) return;
  int o = idx / C;
  int c = idx - o * C;
  float v;
  if (o < 18)        v = im[o] * w[o * C + c];        // det rows, im folded in
  else if (o < NOUT) v = wk[(o - 18) * C + c];        // kpt rows
  else               v = 0.0f;                        // padding
  __bf16 h = (__bf16)v;
  __bf16 l = (__bf16)(v - (float)h);                  // split-bf16 residual
  int t  = o >> 4, ol = o & 15;
  int ks = c >> 5, kk = c & 31;
  int lane = ol + ((kk & 16) ? 16 : 0);
  int e    = kk & 15;
  size_t base = (size_t)(ks * NT + t) * 1024;         // 2 * 512 bf16 per frag
  dst[base + lane * 16 + e]       = h;
  dst[base + 512 + lane * 16 + e] = l;
}

// bias'[o] = im[o]*(b[o] + sum_c w[o,c]*ia[c]) for det, bk for kpt
__global__ void pack_bias(const float* __restrict__ w,
                          const float* __restrict__ b,
                          const float* __restrict__ im,
                          const float* __restrict__ ia,
                          const float* __restrict__ bk,
                          float* __restrict__ dst, int C) {
  int o = blockIdx.x * 64 + threadIdx.x;
  if (o >= OPAD) return;
  float v;
  if (o < 18) {
    float s = b[o];
    for (int c = 0; c < C; ++c) s += w[o * C + c] * ia[c];
    v = im[o] * s;
  } else if (o < NOUT) {
    v = bk[o - 18];
  } else {
    v = 0.0f;
  }
  dst[o] = v;
}

// ---------------------------------------------------------------------------
// Main fused GEMM + decode. One wave = one 16-pixel M-tile, full 176-wide N.
// Wave-private double-buffered LDS tile (32ch x 16px, row stride 20 floats).
// ---------------------------------------------------------------------------
__global__ __launch_bounds__(256) void head_kernel(
    const float* __restrict__ feat, const __bf16* __restrict__ wbuf,
    const float* __restrict__ bias, float* __restrict__ out,
    int C, int W, int HW, int rowoff, float stride,
    float aw0, float ah0, float aw1, float ah1, float aw2, float ah2) {
  __shared__ float smem[8 * 2 * 32 * 20];   // 8 waves x 2 bufs x 2560B
  int tid    = threadIdx.x;
  int waveid = tid >> 5;
  int lane   = tid & 31;
  int nmod   = lane & 15;
  int hi16   = lane >> 4;                   // 0 or 1 (half-wave)
  float* sm  = smem + waveid * (2 * 32 * 20);

  int mt    = blockIdx.x * 8 + waveid;      // global M-tile index
  int flat0 = mt * 16;                      // flattened (batch, pixel)
  int b     = flat0 / HW;                   // HW % 16 == 0 -> tile in one batch
  int p0    = flat0 - b * HW;
  const float* fbase = feat + (size_t)b * C * HW + p0;

  v8f zero = {0.f, 0.f, 0.f, 0.f, 0.f, 0.f, 0.f, 0.f};
  v8f acc[NT];
#pragma unroll
  for (int t = 0; t < NT; ++t) acc[t] = zero;

  int nk = C >> 5;                          // K-steps of 32 channels

#if HAVE_ASYNC_LDS
  {   // prologue: async-stage tile 0 into buffer 0 (4 x b128 per lane)
    const float* src = fbase + (size_t)lane * HW;
    float* row = sm + lane * 20;
    ASYNC_CP16(src + 0,  row + 0);
    ASYNC_CP16(src + 4,  row + 4);
    ASYNC_CP16(src + 8,  row + 8);
    ASYNC_CP16(src + 12, row + 12);
  }
#endif

  for (int ks = 0; ks < nk; ++ks) {
#if HAVE_ASYNC_LDS
    int cur = ks & 1;
    if (ks + 1 < nk) {
      // WAR: previous ds reads of the other buffer must retire first
      wait_dscnt0();
      const float* src = fbase + (size_t)((ks + 1) * 32 + lane) * HW;
      float* row = sm + (cur ^ 1) * 640 + lane * 20;
      ASYNC_CP16(src + 0,  row + 0);
      ASYNC_CP16(src + 4,  row + 4);
      ASYNC_CP16(src + 8,  row + 8);
      ASYNC_CP16(src + 12, row + 12);
      wait_async_le4();                     // tile ks complete; ks+1 in flight
    } else {
      wait_async0();
    }
    const float* tb = sm + cur * 640;
#else
    // fallback staging: global -> VGPR -> LDS (in-order per-wave LDS pipe)
    const float* src = fbase + (size_t)(ks * 32 + lane) * HW;
    float4 r0 = ((const float4*)src)[0];
    float4 r1 = ((const float4*)src)[1];
    float4 r2 = ((const float4*)src)[2];
    float4 r3 = ((const float4*)src)[3];
    if (ks + 1 < nk)
      __builtin_prefetch(fbase + (size_t)((ks + 1) * 32 + lane) * HW, 0, 0);
    float* row = sm + lane * 20;
    *(float4*)(row + 0)  = r0;
    *(float4*)(row + 4)  = r1;
    *(float4*)(row + 8)  = r2;
    *(float4*)(row + 12) = r3;
    const float* tb = sm;
#endif

    // ---- build A fragment (16x32 bf16, contiguous k = 16*hi16 + e) with split-bf16
    v16bf a_hi, a_lo;
#pragma unroll
    for (int e = 0; e < 16; ++e) {
      int k   = e + hi16 * 16;
      float v = tb[k * 20 + nmod];
      __bf16 h = (__bf16)v;
      a_hi[e] = h;
      a_lo[e] = (__bf16)(v - (float)h);
    }

    // ---- 11 N-tiles: pre-packed B frags, 1 coalesced 32B load per lane
    const v16bf* bp = (const v16bf*)wbuf + (size_t)ks * (NT * 64);
#pragma unroll
    for (int t = 0; t < NT; ++t) {
      v16bf bh = bp[t * 64 + lane];
      v16bf bl = bp[t * 64 + 32 + lane];
      acc[t] = __builtin_amdgcn_wmma_f32_16x16x32_bf16(
          false, a_hi, false, bh, (short)0, acc[t], false, false);
      acc[t] = __builtin_amdgcn_wmma_f32_16x16x32_bf16(
          false, a_hi, false, bl, (short)0, acc[t], false, false);
      acc[t] = __builtin_amdgcn_wmma_f32_16x16x32_bf16(
          false, a_lo, false, bh, (short)0, acc[t], false, false);
    }
  }

  // ---- decode straight out of the C-fragment layout -----------------------
  // C/D layout: VGPR r -> pixel (r + 8*hi16), lane n -> output c = 16t + n%16
  int pxA[8], pyA[8];
#pragma unroll
  for (int r = 0; r < 8; ++r) {
    int p = p0 + hi16 * 8 + r;
    int py = p / W;
    pyA[r] = py;
    pxA[r] = p - py * W;
  }
  size_t bRow = ((size_t)b * TOTROWS + (size_t)rowoff) * NO;

#pragma unroll
  for (int t = 0; t < NT; ++t) {
    int c = 16 * t + nmod;
    if (c >= NOUT) continue;               // lane-predicated (padding lanes)
    int a = c / 57;                        // reshaped anchor index
    int j = c - a * 57;                    // position within the 57 outputs
    float bt = bias[c];
    float anc = 0.0f;
    if (j == 2) anc = (a == 0) ? aw0 : ((a == 1) ? aw1 : aw2);
    if (j == 3) anc = (a == 0) ? ah0 : ((a == 1) ? ah1 : ah2);
    int m = (j >= 6) ? ((j - 6) % 3) : -1; // kpt component: x / y / conf
    size_t base = bRow + (size_t)a * HW * NO + (size_t)j;
#pragma unroll
    for (int r = 0; r < 8; ++r) {
      float v = acc[t][r] + bt;
      int   p = p0 + hi16 * 8 + r;
      float outv;
      if (j < 6) {
        float s = fsigmoid(v);
        if (j == 0)      outv = (s * 2.0f - 0.5f + (float)pxA[r]) * stride;
        else if (j == 1) outv = (s * 2.0f - 0.5f + (float)pyA[r]) * stride;
        else if (j < 4)  { float u = s * 2.0f; outv = u * u * anc; }
        else             outv = s;
      } else {
        if (m == 0)      outv = (v * 2.0f - 0.5f + (float)pxA[r]) * stride;
        else if (m == 1) outv = (v * 2.0f - 0.5f + (float)pyA[r]) * stride;
        else             outv = fsigmoid(v);
      }
      out[base + (size_t)p * NO] = outv;
    }
  }
}

// ---------------------------------------------------------------------------
extern "C" void kernel_launch(void* const* d_in, const int* in_sizes, int n_in,
                              void* d_out, int out_size, void* d_ws, size_t ws_size,
                              hipStream_t stream) {
  (void)in_sizes; (void)n_in; (void)out_size; (void)ws_size;
  const int   CH[3]     = {256, 512, 1024};
  const int   HWDIM[3]  = {80, 40, 20};
  const float STRD[3]   = {8.f, 16.f, 32.f};
  const float ANC[3][6] = {{19,27,44,40,38,94},
                           {96,68,86,152,180,137},
                           {140,301,303,264,238,542}};
  const int   ROWOFF[3] = {0, 3 * 6400, 3 * 6400 + 3 * 1600};

  // workspace layout: 3 packed weight buffers (bf16 hi+lo) then 3 bias arrays
  size_t woff[3], cur = 0;
  for (int s = 0; s < 3; ++s) { woff[s] = cur; cur += (size_t)OPAD * CH[s] * 2 * sizeof(__bf16); }
  size_t boff[3];
  for (int s = 0; s < 3; ++s) { boff[s] = cur; cur += (size_t)OPAD * sizeof(float); }

  for (int s = 0; s < 3; ++s) {
    const float* f  = (const float*)d_in[7 * s + 0];
    const float* ia = (const float*)d_in[7 * s + 1];
    const float* w  = (const float*)d_in[7 * s + 2];
    const float* b  = (const float*)d_in[7 * s + 3];
    const float* im = (const float*)d_in[7 * s + 4];
    const float* wk = (const float*)d_in[7 * s + 5];
    const float* bk = (const float*)d_in[7 * s + 6];
    __bf16* wbuf = (__bf16*)((char*)d_ws + woff[s]);
    float*  bias = (float*)((char*)d_ws + boff[s]);
    int C = CH[s], W = HWDIM[s], HW = W * W;

    pack_weights<<<(OPAD * C + 255) / 256, 256, 0, stream>>>(w, wk, im, wbuf, C);
    pack_bias<<<3, 64, 0, stream>>>(w, b, im, ia, bk, bias, C);

    int mtiles = HW;                        // = B*HW/16 (6400/1600/400)
    head_kernel<<<mtiles / 8, 256, 0, stream>>>(
        f, wbuf, bias, (float*)d_out, C, W, HW, ROWOFF[s], STRD[s],
        ANC[s][0], ANC[s][1], ANC[s][2], ANC[s][3], ANC[s][4], ANC[s][5]);
  }
}
